// ByteDecoder_7172595384965
// MI455X (gfx1250) — matrix operand order repaired and checked by reading, not compiled, not run on hardware
//
#include <hip/hip_runtime.h>
#include <hip/hip_bf16.h>
#include <math.h>

#define B_  2
#define S_  1024
#define D_  1024
#define H_  16
#define L_  9
#define P_  128
#define V_  256
#define DH_ 64

typedef __attribute__((ext_vector_type(16))) __bf16 v16bf;
typedef __attribute__((ext_vector_type(4)))  __bf16 v4bf;
typedef __attribute__((ext_vector_type(8)))  float  v8f;
typedef __attribute__((ext_vector_type(4)))  int    v4i;

__device__ __forceinline__ v8f wmma_bf16(v16bf a, v16bf b, v8f c) {
  return __builtin_amdgcn_wmma_f32_16x16x32_bf16(false, a, false, b, (short)0, c, false, false);
}

// Async global->LDS copy of 16 bytes (b128), tracked by ASYNCcnt.
__device__ __forceinline__ void async_copy_b128(const __bf16* g, __bf16* l) {
#if __has_builtin(__builtin_amdgcn_global_load_async_to_lds_b128)
  __builtin_amdgcn_global_load_async_to_lds_b128(
      (__attribute__((address_space(1))) v4i*)g,
      (__attribute__((address_space(3))) v4i*)l, 0, 0);
#else
  unsigned loff = (unsigned)(unsigned long long)(__attribute__((address_space(3))) __bf16*)l;
  asm volatile("global_load_async_to_lds_b128 %0, %1, off"
               :: "v"(loff), "v"(g) : "memory");
#endif
}

__device__ __forceinline__ void wait_async() {
#if __has_builtin(__builtin_amdgcn_s_wait_asynccnt)
  __builtin_amdgcn_s_wait_asynccnt(0);
#else
  asm volatile("s_wait_asynccnt 0x0" ::: "memory");
#endif
}

// Build a 16-element bf16 fragment from two 8-contiguous LDS runs.
__device__ __forceinline__ v16bf ld_frag(const __bf16* lo, const __bf16* hi) {
  v4bf a0 = *(const v4bf*)lo;
  v4bf a1 = *(const v4bf*)(lo + 4);
  v4bf a2 = *(const v4bf*)hi;
  v4bf a3 = *(const v4bf*)(hi + 4);
  v16bf r;
#pragma unroll
  for (int i = 0; i < 4; ++i) { r[i] = a0[i]; r[4 + i] = a1[i]; r[8 + i] = a2[i]; r[12 + i] = a3[i]; }
  return r;
}

// ---------------------------------------------------------------------------
// GEMM: Y[M,N] = X[M,K] @ W[N,K]^T + bias. X,W are bf16; accum f32.
// Output f32 or bf16, optional exact-GELU. Double-buffered async LDS tiles.
// M%128==0, N%128==0, K%32==0 at all call sites.
// ---------------------------------------------------------------------------
#define GS 40  // LDS row stride in bf16: 80B rows -> 16B-aligned segs, conflict-free

__global__ __launch_bounds__(256)
void gemm_kernel(void* __restrict__ Yv, const __bf16* __restrict__ X,
                 const __bf16* __restrict__ Wt, const float* __restrict__ bias,
                 int M, int N, int K, int act, int outBf16) {
  __shared__ __attribute__((aligned(16))) __bf16 Xs[2][128][GS];
  __shared__ __attribute__((aligned(16))) __bf16 Ws[2][128][GS];
  const int tid  = threadIdx.x;
  const int lane = tid & 31;
  const int wave = tid >> 5;
  const int mBase = blockIdx.y * 128;
  const int nBase = blockIdx.x * 128;
  const int half  = lane >> 4;
  const int l15   = lane & 15;
  const int ak    = half * 8;
  const int bk    = half * 8;

  v8f c[8] = {};

  // 512 b128 segments per 128x32 tile; 2 per thread for X and for W
  const int seg0row = tid >> 2;
  const int seg0off = (tid & 3) * 8;
  const int seg1row = (tid + 256) >> 2;
  const int seg1off = ((tid + 256) & 3) * 8;

  auto stage = [&](int buf, int kb) {
    async_copy_b128(X + (size_t)(mBase + seg0row) * K + kb + seg0off, &Xs[buf][seg0row][seg0off]);
    async_copy_b128(X + (size_t)(mBase + seg1row) * K + kb + seg1off, &Xs[buf][seg1row][seg1off]);
    async_copy_b128(Wt + (size_t)(nBase + seg0row) * K + kb + seg0off, &Ws[buf][seg0row][seg0off]);
    async_copy_b128(Wt + (size_t)(nBase + seg1row) * K + kb + seg1off, &Ws[buf][seg1row][seg1off]);
  };

  stage(0, 0);
  wait_async();
  __syncthreads();

  int buf = 0;
  for (int kb = 0; kb < K; kb += 32) {
    if (kb + 32 < K) stage(buf ^ 1, kb + 32);

    const int arow = wave * 16 + l15;
    v16bf a = ld_frag(&Xs[buf][arow][ak], &Xs[buf][arow][ak + 16]);
    v16bf bf[8];
#pragma unroll
    for (int nt = 0; nt < 8; ++nt)
      bf[nt] = ld_frag(&Ws[buf][nt * 16 + l15][bk], &Ws[buf][nt * 16 + l15][bk + 16]);
#pragma unroll
    for (int nt = 0; nt < 8; ++nt) c[nt] = wmma_bf16(a, bf[nt], c[nt]);

    wait_async();
    __syncthreads();
    buf ^= 1;
  }

  // Epilogue: C layout -> row m = r + 8*half, col n = l15 per 16x16 tile
  const int mOff = wave * 16 + half * 8;
#pragma unroll
  for (int nt = 0; nt < 8; ++nt) {
    const int n = nBase + nt * 16 + l15;
    const float bv = bias ? bias[n] : 0.f;
#pragma unroll
    for (int r = 0; r < 8; ++r) {
      float v = c[nt][r] + bv;
      if (act == 1) v = 0.5f * v * (1.0f + erff(v * 0.70710678118654752f));
      const size_t idx = (size_t)(mBase + mOff + r) * N + n;
      if (outBf16) ((__bf16*)Yv)[idx] = (__bf16)v;
      else         ((float*)Yv)[idx] = v;
    }
  }
}

// ---------------------------------------------------------------------------
// Fused flash attention, bf16 in / bf16 out, f32 softmax + accum.
// block = (b,h) x 128 q rows, 8 waves x 16 q rows.
// mode 0: causal (k > q masked). mode 1: cross (patch p > pidx[b,q] masked).
// ---------------------------------------------------------------------------
#define KS_STRIDE 72  // 144B rows: async-b128 dst 16B-aligned, conflict-free
#define VT_STRIDE 36
#define PS_STRIDE 36

__global__ __launch_bounds__(256)
void attn_kernel(__bf16* __restrict__ O, const __bf16* __restrict__ Qm,
                 const __bf16* __restrict__ Km, const __bf16* __restrict__ Vm,
                 const int* __restrict__ pidx,
                 int Sq, int Skv, int qStride, int kStride, int vStride, int mode) {
  __shared__ __attribute__((aligned(16))) __bf16 Ks[32][KS_STRIDE];  // [kv][dh]
  __shared__ __attribute__((aligned(16))) __bf16 Vt[DH_][VT_STRIDE]; // [dh][kv]
  __shared__ __attribute__((aligned(16))) __bf16 Ps[8][16][PS_STRIDE];

  const int tid  = threadIdx.x;
  const int lane = tid & 31;
  const int wave = tid >> 5;
  const int b = blockIdx.y / H_;
  const int h = blockIdx.y % H_;
  const int qTile = blockIdx.x * 128;
  const int qWave = qTile + wave * 16;
  const int half  = lane >> 4;
  const int l15   = lane & 15;
  const int ak    = half * 8;
  const int bkk   = half * 8;

  // Q fragments for dh 0..31 / 32..63 straight from global bf16
  const __bf16* qr = Qm + (size_t)(b * Sq + qWave + l15) * qStride + h * DH_;
  v16bf aq0 = ld_frag(qr + ak, qr + ak + 16);
  v16bf aq1 = ld_frag(qr + 32 + ak, qr + 32 + ak + 16);

  int qrow[8], plim[8];
#pragma unroll
  for (int r = 0; r < 8; ++r) {
    qrow[r] = qWave + half * 8 + r;
    plim[r] = (mode == 1) ? pidx[b * Sq + qrow[r]] : qrow[r];
  }

  v8f o[4] = {};
  float mrun[8], lrun[8];
#pragma unroll
  for (int r = 0; r < 8; ++r) { mrun[r] = -1e30f; lrun[r] = 0.f; }

  int kvEnd = Skv;
  if (mode == 0) kvEnd = min(Skv, qTile + 128);

  // K-tile async loader: 256 b128 segs (1/thread): row=tid>>3, off=(tid&7)*8
  const int krow = tid >> 3;
  const int koff = (tid & 7) * 8;

  for (int kv = 0; kv < kvEnd; kv += 32) {
    async_copy_b128(Km + (size_t)(b * Skv + kv + krow) * kStride + h * DH_ + koff,
                    &Ks[krow][koff]);
    // V tile: load bf16 + transpose into LDS
#pragma unroll
    for (int it = 0; it < 2; ++it) {
      const int idx = tid + it * 256;
      const int row = idx >> 4;
      const int c4  = (idx & 15) * 4;
      v4bf vf = *(const v4bf*)(Vm + (size_t)(b * Skv + kv + row) * vStride + h * DH_ + c4);
      Vt[c4 + 0][row] = vf[0];
      Vt[c4 + 1][row] = vf[1];
      Vt[c4 + 2][row] = vf[2];
      Vt[c4 + 3][row] = vf[3];
    }
    wait_async();
    __syncthreads();

    // scores: two 16-kv-col tiles, K-dim = dh (two 32-chunks)
    v8f s[2];
#pragma unroll
    for (int nt = 0; nt < 2; ++nt) {
      const int brow = nt * 16 + l15;
      v16bf b0 = ld_frag(&Ks[brow][bkk], &Ks[brow][bkk + 16]);
      v16bf b1 = ld_frag(&Ks[brow][32 + bkk], &Ks[brow][32 + bkk + 16]);
      v8f acc = {};
      acc = wmma_bf16(aq0, b0, acc);
      acc = wmma_bf16(aq1, b1, acc);
      s[nt] = acc;
    }

    // scale + mask
#pragma unroll
    for (int nt = 0; nt < 2; ++nt)
#pragma unroll
      for (int r = 0; r < 8; ++r) {
        const int kcol = kv + nt * 16 + l15;
        float v = s[nt][r] * 0.125f;
        if (kcol > plim[r]) v = -1e9f;
        s[nt][r] = v;
      }

    // online softmax: row reductions within 16-lane groups
    float mnew[8], alpha[8];
#pragma unroll
    for (int r = 0; r < 8; ++r) {
      float mx = fmaxf(s[0][r], s[1][r]);
#pragma unroll
      for (int d = 1; d < 16; d <<= 1) mx = fmaxf(mx, __shfl_xor(mx, d, 32));
      mnew[r]  = fmaxf(mrun[r], mx);
      alpha[r] = __expf(mrun[r] - mnew[r]);
    }
#pragma unroll
    for (int r = 0; r < 8; ++r) {
      float p0 = __expf(s[0][r] - mnew[r]);
      float p1 = __expf(s[1][r] - mnew[r]);
      s[0][r] = p0; s[1][r] = p1;
      float sm = p0 + p1;
#pragma unroll
      for (int d = 1; d < 16; d <<= 1) sm += __shfl_xor(sm, d, 32);
      lrun[r] = lrun[r] * alpha[r] + sm;
      mrun[r] = mnew[r];
    }
#pragma unroll
    for (int dt = 0; dt < 4; ++dt)
#pragma unroll
      for (int r = 0; r < 8; ++r) o[dt][r] *= alpha[r];

    // restage P (C layout) into A-fragment layout via per-wave LDS
#pragma unroll
    for (int r = 0; r < 8; ++r) {
      Ps[wave][half * 8 + r][l15]      = (__bf16)s[0][r];
      Ps[wave][half * 8 + r][16 + l15] = (__bf16)s[1][r];
    }
    v16bf ap = ld_frag(&Ps[wave][l15][ak], &Ps[wave][l15][ak + 16]);

    // O += P @ V : four 16-dh-col tiles
#pragma unroll
    for (int dt = 0; dt < 4; ++dt) {
      const int vrow = dt * 16 + l15;
      v16bf bv = ld_frag(&Vt[vrow][bkk], &Vt[vrow][bkk + 16]);
      o[dt] = wmma_bf16(ap, bv, o[dt]);
    }
    __syncthreads();
  }

  // normalize + store bf16: O[b, q, h*64 + dh]
#pragma unroll
  for (int r = 0; r < 8; ++r) {
    const float inv = 1.0f / lrun[r];
    const size_t base = (size_t)(b * Sq + qrow[r]) * D_ + h * DH_;
#pragma unroll
    for (int dt = 0; dt < 4; ++dt) O[base + dt * 16 + l15] = (__bf16)(o[dt][r] * inv);
  }
}

// ---------------------------------------------------------------------------
__global__ __launch_bounds__(256)
void layernorm_kernel(__bf16* __restrict__ out, const float* __restrict__ in,
                      const float* __restrict__ g, const float* __restrict__ bt) {
  __shared__ float red1[256], red2[256];
  const int row = blockIdx.x;
  const int tid = threadIdx.x;
  const float* x = in + (size_t)row * D_;
  float s1 = 0.f, s2 = 0.f;
#pragma unroll
  for (int i = 0; i < 4; ++i) {
    float v = x[tid + i * 256];
    s1 += v; s2 += v * v;
  }
  red1[tid] = s1; red2[tid] = s2;
  __syncthreads();
  for (int off = 128; off > 0; off >>= 1) {
    if (tid < off) { red1[tid] += red1[tid + off]; red2[tid] += red2[tid + off]; }
    __syncthreads();
  }
  const float mean = red1[0] * (1.0f / D_);
  const float var  = red2[0] * (1.0f / D_) - mean * mean;
  const float rs   = rsqrtf(var + 1e-5f);
  __bf16* y = out + (size_t)row * D_;
#pragma unroll
  for (int i = 0; i < 4; ++i) {
    const int idx = tid + i * 256;
    y[idx] = (__bf16)((x[idx] - mean) * rs * g[idx] + bt[idx]);
  }
}

__global__ __launch_bounds__(256)
void embed_kernel(float* __restrict__ x, const int* __restrict__ seq,
                  const float* __restrict__ bemb, const float* __restrict__ pemb) {
  const int row = blockIdx.x;  // b*S + s
  const int s = row % S_;
  const int byte = seq[row];
  const int tid = threadIdx.x;
#pragma unroll
  for (int i = 0; i < 4; ++i) {
    const int idx = tid + i * 256;
    x[(size_t)row * D_ + idx] = bemb[(size_t)byte * D_ + idx] + pemb[(size_t)s * D_ + idx];
  }
}

__global__ void pidx_kernel(int* __restrict__ pidx, const int* __restrict__ pb) {
  const int b = threadIdx.x;
  if (b >= B_) return;
  int acc = 0;
  for (int s = 0; s < S_; ++s) { acc += pb[b * S_ + s]; pidx[b * S_ + s] = acc; }
}

__global__ __launch_bounds__(256)
void add_kernel(float* __restrict__ x, const float* __restrict__ y, int n) {
  const int i = blockIdx.x * 256 + threadIdx.x;
  if (i < n) x[i] += y[i];
}

// f32 -> bf16 streaming conversion (n % 8 == 0)
__global__ __launch_bounds__(256)
void cvt_bf16_kernel(__bf16* __restrict__ out, const float* __restrict__ in, int n) {
  const int i = (blockIdx.x * 256 + threadIdx.x) * 8;
  if (i < n) {
    float4 f0 = *(const float4*)(in + i);
    float4 f1 = *(const float4*)(in + i + 4);
    v4bf t0, t1;
    t0[0] = (__bf16)f0.x; t0[1] = (__bf16)f0.y; t0[2] = (__bf16)f0.z; t0[3] = (__bf16)f0.w;
    t1[0] = (__bf16)f1.x; t1[1] = (__bf16)f1.y; t1[2] = (__bf16)f1.z; t1[3] = (__bf16)f1.w;
    *(v4bf*)(out + i) = t0;
    *(v4bf*)(out + i + 4) = t1;
  }
}

// ---------------------------------------------------------------------------
extern "C" void kernel_launch(void* const* d_in, const int* in_sizes, int n_in,
                              void* d_out, int out_size, void* d_ws, size_t ws_size,
                              hipStream_t stream) {
  const int*   byte_seq = (const int*)d_in[0];
  const float* patches  = (const float*)d_in[1];
  const int*   pbound   = (const int*)d_in[2];
  const float* byte_emb = (const float*)d_in[3];
  const float* pos_emb  = (const float*)d_in[4];
  const float* sa_in_w  = (const float*)d_in[5];
  const float* sa_in_b  = (const float*)d_in[6];
  const float* sa_out_w = (const float*)d_in[7];
  const float* sa_out_b = (const float*)d_in[8];
  const float* ca_in_w  = (const float*)d_in[9];
  const float* ca_in_b  = (const float*)d_in[10];
  const float* ca_out_w = (const float*)d_in[11];
  const float* ca_out_b = (const float*)d_in[12];
  const float* ffn_w1   = (const float*)d_in[13];
  const float* ffn_b1   = (const float*)d_in[14];
  const float* ffn_w2   = (const float*)d_in[15];
  const float* ffn_b2   = (const float*)d_in[16];
  const float* ln_g     = (const float*)d_in[17];
  const float* ln_b     = (const float*)d_in[18];
  const float* norm_g   = (const float*)d_in[19];
  const float* norm_b   = (const float*)d_in[20];
  const float* out_w    = (const float*)d_in[21];

  const int M  = B_ * S_;     // 2048 token rows
  const int DD = D_ * D_;

  float*  x     = (float*)d_ws;
  float*  proj  = x + (size_t)M * D_;
  __bf16* h     = (__bf16*)(proj + (size_t)M * D_);
  __bf16* qkv   = h     + (size_t)M * D_;
  __bf16* kvp   = qkv   + (size_t)M * 3 * D_;
  __bf16* attno = kvp   + (size_t)B_ * P_ * 2 * D_;
  __bf16* ffn1  = attno + (size_t)M * D_;
  __bf16* pconv = ffn1  + (size_t)M * 4 * D_;
  __bf16* wst   = pconv + (size_t)B_ * P_ * D_;   // 16*DD per-layer staging
  __bf16* wout  = wst   + (size_t)16 * DD;
  int*    pidx  = (int*)(wout + (size_t)V_ * D_);

  __bf16* w_sa_in  = wst;
  __bf16* w_sa_out = w_sa_in  + (size_t)3 * DD;
  __bf16* w_ca_in  = w_sa_out + (size_t)1 * DD;
  __bf16* w_ca_out = w_ca_in  + (size_t)3 * DD;
  __bf16* w_f1     = w_ca_out + (size_t)1 * DD;
  __bf16* w_f2     = w_f1     + (size_t)4 * DD;

  dim3 blk(256);
  auto cvt = [&](__bf16* dst, const float* src, size_t n) {
    cvt_bf16_kernel<<<(unsigned)(n / 2048), blk, 0, stream>>>(dst, src, (int)n);
  };
  auto gemm = [&](void* Y, const __bf16* X, const __bf16* Wm, const float* bias,
                  int MM, int NN, int KK, int act, int outBf16) {
    dim3 g(NN / 128, MM / 128);
    gemm_kernel<<<g, blk, 0, stream>>>(Y, X, Wm, bias, MM, NN, KK, act, outBf16);
  };

  embed_kernel<<<M, blk, 0, stream>>>(x, byte_seq, byte_emb, pos_emb);
  pidx_kernel<<<1, 32, 0, stream>>>(pidx, pbound);
  cvt(pconv, patches, (size_t)B_ * P_ * D_);
  cvt(wout, out_w, (size_t)V_ * D_);

  const int nTot = M * D_;
  for (int l = 0; l < L_; ++l) {
    const float* lg = ln_g + (size_t)l * 3 * D_;
    const float* lb = ln_b + (size_t)l * 3 * D_;

    // stage this layer's weights as bf16
    cvt(w_sa_in,  sa_in_w  + (size_t)l * 3 * DD, (size_t)3 * DD);
    cvt(w_sa_out, sa_out_w + (size_t)l * DD,     (size_t)1 * DD);
    cvt(w_ca_in,  ca_in_w  + (size_t)l * 3 * DD, (size_t)3 * DD);
    cvt(w_ca_out, ca_out_w + (size_t)l * DD,     (size_t)1 * DD);
    cvt(w_f1,     ffn_w1   + (size_t)l * 4 * DD, (size_t)4 * DD);
    cvt(w_f2,     ffn_w2   + (size_t)l * 4 * DD, (size_t)4 * DD);

    // --- self attention ---
    layernorm_kernel<<<M, blk, 0, stream>>>(h, x, lg, lb);
    gemm(qkv, h, w_sa_in, sa_in_b + (size_t)l * 3 * D_, M, 3 * D_, D_, 0, 1);
    {
      dim3 g(S_ / 128, B_ * H_);
      attn_kernel<<<g, blk, 0, stream>>>(attno, qkv, qkv + D_, qkv + 2 * D_,
                                         nullptr, S_, S_, 3 * D_, 3 * D_, 3 * D_, 0);
    }
    gemm(proj, attno, w_sa_out, sa_out_b + (size_t)l * D_, M, D_, D_, 0, 0);
    add_kernel<<<(nTot + 255) / 256, blk, 0, stream>>>(x, proj, nTot);

    // --- cross attention ---
    layernorm_kernel<<<M, blk, 0, stream>>>(h, x, lg + D_, lb + D_);
    gemm(qkv, h, w_ca_in, ca_in_b + (size_t)l * 3 * D_, M, D_, D_, 0, 1);
    gemm(kvp, pconv, w_ca_in + (size_t)DD, ca_in_b + (size_t)l * 3 * D_ + D_,
         B_ * P_, 2 * D_, D_, 0, 1);
    {
      dim3 g(S_ / 128, B_ * H_);
      attn_kernel<<<g, blk, 0, stream>>>(attno, qkv, kvp, kvp + D_,
                                         pidx, S_, P_, D_, 2 * D_, 2 * D_, 1);
    }
    gemm(proj, attno, w_ca_out, ca_out_b + (size_t)l * D_, M, D_, D_, 0, 0);
    add_kernel<<<(nTot + 255) / 256, blk, 0, stream>>>(x, proj, nTot);

    // --- FFN ---
    layernorm_kernel<<<M, blk, 0, stream>>>(h, x, lg + 2 * D_, lb + 2 * D_);
    gemm(ffn1, h, w_f1, ffn_b1 + (size_t)l * 4 * D_, M, 4 * D_, D_, 1, 1);
    gemm(proj, ffn1, w_f2, ffn_b2 + (size_t)l * D_, M, D_, 4 * D_, 0, 0);
    add_kernel<<<(nTot + 255) / 256, blk, 0, stream>>>(x, proj, nTot);
  }

  layernorm_kernel<<<M, blk, 0, stream>>>(h, x, norm_g, norm_b);
  gemm(d_out, h, wout, nullptr, M, V_, D_, 0, 0);
}
